// MultiHeadAttention_45552423141993
// MI455X (gfx1250) — matrix operand order repaired and checked by reading, not compiled, not run on hardware
//
#include <hip/hip_runtime.h>

// ---------------------------------------------------------------- constants
#define SEQ    4096
#define DMODEL 768
#define NH     12
#define DH     64
#define SCALE  0.125f        // 1/sqrt(64)

typedef __attribute__((ext_vector_type(16))) _Float16 v16h;
typedef __attribute__((ext_vector_type(8)))  float    v8f;
typedef __attribute__((ext_vector_type(4)))  int      v4i;

#define AS1 __attribute__((address_space(1)))
#define AS3 __attribute__((address_space(3)))

// -------------------------------------------------- async global->LDS copy
// CDNA5 GLOBAL_LOAD_ASYNC_TO_LDS_B128 (ASYNCcnt-tracked, no VGPR data path).
// Guarded: falls back to a plain 128-bit copy if the builtin is absent.
__device__ __forceinline__ void async_copy_b128(const _Float16* g, _Float16* s) {
#if __has_builtin(__builtin_amdgcn_global_load_async_to_lds_b128)
    __builtin_amdgcn_global_load_async_to_lds_b128((AS1 v4i*)g, (AS3 v4i*)s,
                                                   /*offset=*/0, /*cpol=*/0);
#else
    *(int4*)s = *(const int4*)g;
#endif
}

__device__ __forceinline__ void wait_async0() {
#if __has_builtin(__builtin_amdgcn_s_wait_asynccnt)
    __builtin_amdgcn_s_wait_asynccnt(0);
#else
    asm volatile("s_wait_asynccnt 0x0" ::: "memory");
#endif
}

// ------------------------------------------------------- WMMA frag helpers
// wave32 16x16x32 f16 layouts used here (A/B in this code's contiguous k order, not ISA table order):
//   A (16x32, MxK): lane l: M=l&15, half=l>>4. VGPR v holds K pair at
//       base = 16*half + 2v  -> one contiguous 16-half run (matches B)
//   B (32x16, KxN): lane l: N=l&15, half=l>>4, element i -> K = 16*half + i
//   C/D (16x16 f32): lane l: N=l&15; VGPR v -> M = v + 8*(l>>4)

__device__ __forceinline__ int a_pair_k(int v, int half) {
    int base = 2 * v;
    return base + 16 * half;
}

// A fragment from an f16 row-major [rows][ld] source
__device__ __forceinline__ v16h load_a_h16(const _Float16* __restrict__ src,
                                           int ld, int m0, int lane) {
    int M = lane & 15, half = lane >> 4;
    const _Float16* row = src + (size_t)(m0 + M) * ld;
    v16h a;
#pragma unroll
    for (int v = 0; v < 8; ++v) {
        int k = a_pair_k(v, half);
        a[2 * v]     = row[k];
        a[2 * v + 1] = row[k + 1];
    }
    return a;
}

// A fragment from an f32 row-major source, converted to f16 on the fly
__device__ __forceinline__ v16h load_a_f32(const float* __restrict__ src,
                                           size_t ld, int m0, int lane) {
    int M = lane & 15, half = lane >> 4;
    const float* row = src + (size_t)(m0 + M) * ld;
    v16h a;
#pragma unroll
    for (int v = 0; v < 8; ++v) {
        int k = a_pair_k(v, half);
        a[2 * v]     = (_Float16)row[k];
        a[2 * v + 1] = (_Float16)row[k + 1];
    }
    return a;
}

// B fragment from a k-major [k][ld] f16 source (element (k,n) at src[k*ld+n])
__device__ __forceinline__ v16h load_b_kmajor(const _Float16* __restrict__ src,
                                              int ld, int n0, int lane) {
    int n = n0 + (lane & 15), half = lane >> 4;
    v16h b;
#pragma unroll
    for (int i = 0; i < 16; ++i) b[i] = src[(size_t)(16 * half + i) * ld + n];
    return b;
}

// B fragment from an n-major [n][ld] f16 source (element (k,n) at src[n*ld+k])
__device__ __forceinline__ v16h load_b_nmajor(const _Float16* __restrict__ src,
                                              int ld, int n0, int lane) {
    int n = n0 + (lane & 15), half = lane >> 4;
    const _Float16* col = src + (size_t)n * ld + 16 * half;
    v16h b;
#pragma unroll
    for (int i = 0; i < 16; ++i) b[i] = col[i];
    return b;
}

__device__ __forceinline__ v8f wmma_f16(v16h a, v16h b, v8f c) {
    return __builtin_amdgcn_wmma_f32_16x16x32_f16(
        /*neg_a=*/false, a, /*neg_b=*/false, b,
        /*c_mod=*/(short)0, c, /*reuse_a=*/false, /*reuse_b=*/false);
}

// ---------------------------------------------------------------- kernel 1
// Out_h[h][s][dh] (f16) = (X[s][:] @ W[:][:] + bias)[s][h*64+dh]
__global__ __launch_bounds__(128)
void qkv_proj_kernel(const float* __restrict__ X, const float* __restrict__ W,
                     const float* __restrict__ bias, _Float16* __restrict__ out) {
    __shared__ __align__(16) _Float16 As[64 * 32];   // [m][k]
    __shared__ __align__(16) _Float16 BsT[64 * 32];  // [n][k]
    const int m0 = blockIdx.x * 64;
    const int n0 = blockIdx.y * 64;
    const int tid  = threadIdx.x;
    const int lane = tid & 31;
    const int w    = tid >> 5;
    const int wm   = (w >> 1) * 32;
    const int wn   = (w & 1) * 32;

    v8f acc[2][2] = {};
    for (int k0 = 0; k0 < DMODEL; k0 += 32) {
        {   // stage A: 64x32, f32->f16, contiguous global reads
            int r  = tid >> 1;
            int kk = (tid & 1) * 16;
            const float* g = X + (size_t)(m0 + r) * DMODEL + k0 + kk;
            _Float16* s = As + r * 32 + kk;
#pragma unroll
            for (int j = 0; j < 16; ++j) s[j] = (_Float16)g[j];
        }
        {   // stage B transposed: BsT[n][k] = W[k0+k][n0+n]
            int kk = tid >> 2;
            int nn = (tid & 3) * 16;
            const float* g = W + (size_t)(k0 + kk) * DMODEL + n0 + nn;
#pragma unroll
            for (int j = 0; j < 16; ++j)
                BsT[(size_t)(nn + j) * 32 + kk] = (_Float16)g[j];
        }
        __syncthreads();
        v16h af[2], bf[2];
#pragma unroll
        for (int mi = 0; mi < 2; ++mi) af[mi] = load_a_h16(As, 32, wm + mi * 16, lane);
#pragma unroll
        for (int ni = 0; ni < 2; ++ni) bf[ni] = load_b_nmajor(BsT, 32, wn + ni * 16, lane);
#pragma unroll
        for (int mi = 0; mi < 2; ++mi)
#pragma unroll
            for (int ni = 0; ni < 2; ++ni)
                acc[mi][ni] = wmma_f16(af[mi], bf[ni], acc[mi][ni]);
        __syncthreads();
    }
    const int half = lane >> 4, nl = lane & 15;
#pragma unroll
    for (int mi = 0; mi < 2; ++mi)
#pragma unroll
        for (int ni = 0; ni < 2; ++ni) {
            int ng = n0 + wn + ni * 16 + nl;
            int h  = ng >> 6;
            int dd = ng & 63;
            float bb = bias[ng];
#pragma unroll
            for (int v = 0; v < 8; ++v) {
                int mg = m0 + wm + mi * 16 + v + 8 * half;
                out[(size_t)h * SEQ * DH + (size_t)mg * DH + dd] =
                    (_Float16)(acc[mi][ni][v] + bb);
            }
        }
}

// ---------------------------------------------------------------- kernel 2
// Causal logits: L[h][r][c] = scale * Qh[r]·Kh[c]  (c<=r), else -1e9.
// Fully masked 64x64 blocks are skipped (softmax writes the zeros).
__global__ __launch_bounds__(128)
void logits_kernel(const _Float16* __restrict__ qh, const _Float16* __restrict__ kh,
                   float* __restrict__ attn) {
    const int h  = blockIdx.z;
    const int m0 = blockIdx.y * 64;
    const int n0 = blockIdx.x * 64;
    if (n0 > m0 + 63) return;  // uniform per block: entire tile masked
    const int tid  = threadIdx.x;
    const int lane = tid & 31;
    const int w    = tid >> 5;
    const int wm   = (w >> 1) * 32;
    const int wn   = (w & 1) * 32;
    const _Float16* Q = qh + (size_t)h * SEQ * DH;
    const _Float16* K = kh + (size_t)h * SEQ * DH;

    v8f acc[2][2] = {};
#pragma unroll
    for (int k0 = 0; k0 < DH; k0 += 32) {
        v16h af[2], bf[2];
#pragma unroll
        for (int mi = 0; mi < 2; ++mi)
            af[mi] = load_a_h16(Q + k0, DH, m0 + wm + mi * 16, lane);
#pragma unroll
        for (int ni = 0; ni < 2; ++ni)
            bf[ni] = load_b_nmajor(K + k0, DH, n0 + wn + ni * 16, lane);
#pragma unroll
        for (int mi = 0; mi < 2; ++mi)
#pragma unroll
            for (int ni = 0; ni < 2; ++ni)
                acc[mi][ni] = wmma_f16(af[mi], bf[ni], acc[mi][ni]);
    }
    float* L = attn + (size_t)h * SEQ * SEQ;
    const int half = lane >> 4, nl = lane & 15;
#pragma unroll
    for (int mi = 0; mi < 2; ++mi)
#pragma unroll
        for (int ni = 0; ni < 2; ++ni) {
            int c = n0 + wn + ni * 16 + nl;
#pragma unroll
            for (int v = 0; v < 8; ++v) {
                int r = m0 + wm + mi * 16 + v + 8 * half;
                float val = (c <= r) ? acc[mi][ni][v] * SCALE : -1.0e9f;
                L[(size_t)r * SEQ + c] = val;
            }
        }
}

// ---------------------------------------------------------------- kernel 3
// Row softmax over [0..r]; columns > r get exact 0 (reference: exp(-1e9-max)).
__global__ __launch_bounds__(256)
void softmax_kernel(float* __restrict__ attn) {
    const int h = blockIdx.y;
    const int r = blockIdx.x;
    float* row = attn + (size_t)h * SEQ * SEQ + (size_t)r * SEQ;
    const int tid = threadIdx.x;
    const int n_valid = r + 1;
    __shared__ float red[256];

    float m = -3.4e38f;
    for (int j = tid; j < n_valid; j += 256) m = fmaxf(m, row[j]);
    red[tid] = m;
    __syncthreads();
    for (int s = 128; s > 0; s >>= 1) {
        if (tid < s) red[tid] = fmaxf(red[tid], red[tid + s]);
        __syncthreads();
    }
    m = red[0];
    __syncthreads();

    float sum = 0.f;
    for (int j = tid; j < n_valid; j += 256) sum += __expf(row[j] - m);
    red[tid] = sum;
    __syncthreads();
    for (int s = 128; s > 0; s >>= 1) {
        if (tid < s) red[tid] += red[tid + s];
        __syncthreads();
    }
    const float inv = 1.f / red[0];

    for (int j = tid; j < n_valid; j += 256) row[j] = __expf(row[j] - m) * inv;
    for (int j = n_valid + tid; j < SEQ; j += 256) row[j] = 0.f;
}

// ---------------------------------------------------------------- kernel 4
// ctx[s][h*64+n] = sum_k P[h][s][k] * Vh[h][k][n]; causal => k < m0+64 only.
// V tiles are contiguous 4KB blocks -> double-buffered async global->LDS.
__global__ __launch_bounds__(128)
void pv_kernel(const float* __restrict__ attn, const _Float16* __restrict__ vh,
               _Float16* __restrict__ ctx) {
    const int h  = blockIdx.z;
    const int m0 = blockIdx.x * 64;
    const int tid  = threadIdx.x;
    const int lane = tid & 31;
    const int w    = tid >> 5;
    const int wm   = (w >> 1) * 32;
    const int wn   = (w & 1) * 32;
    __shared__ __align__(16) _Float16 Vs[2][32 * DH];  // ping-pong [k][64]
    const float*    P = attn + (size_t)h * SEQ * SEQ;
    const _Float16* V = vh + (size_t)h * SEQ * DH;
    const int kmax = m0 + 64;  // probs are exactly 0 beyond the block's rows

    {   // prologue: async-stage tile 0 (4KB contiguous), 32B per thread
        const _Float16* g = V + tid * 16;
        _Float16*       s = &Vs[0][tid * 16];
        async_copy_b128(g, s);
        async_copy_b128(g + 8, s + 8);
    }

    v8f acc[2][2] = {};
    int buf = 0;
    for (int k0 = 0; k0 < kmax; k0 += 32) {
        wait_async0();       // our async writes into Vs[buf] landed
        __syncthreads();     // everyone's landed -> tile[buf] readable
        if (k0 + 32 < kmax) {   // async-stage next tile into other buffer
            const _Float16* g = V + (size_t)(k0 + 32) * DH + tid * 16;
            _Float16*       s = &Vs[buf ^ 1][tid * 16];
            async_copy_b128(g, s);
            async_copy_b128(g + 8, s + 8);
        }
        v16h af[2], bf[2];
#pragma unroll
        for (int mi = 0; mi < 2; ++mi)
            af[mi] = load_a_f32(P + k0, (size_t)SEQ, m0 + wm + mi * 16, lane);
#pragma unroll
        for (int ni = 0; ni < 2; ++ni)
            bf[ni] = load_b_kmajor(Vs[buf], DH, wn + ni * 16, lane);
#pragma unroll
        for (int mi = 0; mi < 2; ++mi)
#pragma unroll
            for (int ni = 0; ni < 2; ++ni)
                acc[mi][ni] = wmma_f16(af[mi], bf[ni], acc[mi][ni]);
        buf ^= 1;
    }
    const int half = lane >> 4, nl = lane & 15;
#pragma unroll
    for (int mi = 0; mi < 2; ++mi)
#pragma unroll
        for (int ni = 0; ni < 2; ++ni) {
            int n = wn + ni * 16 + nl;
#pragma unroll
            for (int v = 0; v < 8; ++v) {
                int mg = m0 + wm + mi * 16 + v + 8 * half;
                ctx[(size_t)mg * DMODEL + h * DH + n] = (_Float16)acc[mi][ni][v];
            }
        }
}

// ---------------------------------------------------------------- kernel 5
// out[s][n] (f32) = ctx[s][:] (f16) @ Wc[:][n] (f32->f16) + bias[n]
// A tile is a pure f16 copy -> async global->LDS, overlapped with B convert.
__global__ __launch_bounds__(128)
void out_proj_kernel(const _Float16* __restrict__ Xh, const float* __restrict__ W,
                     const float* __restrict__ bias, float* __restrict__ out) {
    __shared__ __align__(16) _Float16 As[64 * 32];
    __shared__ __align__(16) _Float16 BsT[64 * 32];
    const int m0 = blockIdx.x * 64;
    const int n0 = blockIdx.y * 64;
    const int tid  = threadIdx.x;
    const int lane = tid & 31;
    const int w    = tid >> 5;
    const int wm   = (w >> 1) * 32;
    const int wn   = (w & 1) * 32;

    v8f acc[2][2] = {};
    for (int k0 = 0; k0 < DMODEL; k0 += 32) {
        {   // async stage A: 64 rows x 32 f16, 32B per thread
            int r  = tid >> 1;
            int kk = (tid & 1) * 16;
            const _Float16* g = Xh + (size_t)(m0 + r) * DMODEL + k0 + kk;
            _Float16*       s = As + r * 32 + kk;
            async_copy_b128(g, s);
            async_copy_b128(g + 8, s + 8);
        }
        {   // manual stage B transposed with f32->f16 convert
            int kk = tid >> 2;
            int nn = (tid & 3) * 16;
            const float* g = W + (size_t)(k0 + kk) * DMODEL + n0 + nn;
#pragma unroll
            for (int j = 0; j < 16; ++j)
                BsT[(size_t)(nn + j) * 32 + kk] = (_Float16)g[j];
        }
        wait_async0();
        __syncthreads();
        v16h af[2], bf[2];
#pragma unroll
        for (int mi = 0; mi < 2; ++mi) af[mi] = load_a_h16(As, 32, wm + mi * 16, lane);
#pragma unroll
        for (int ni = 0; ni < 2; ++ni) bf[ni] = load_b_nmajor(BsT, 32, wn + ni * 16, lane);
#pragma unroll
        for (int mi = 0; mi < 2; ++mi)
#pragma unroll
            for (int ni = 0; ni < 2; ++ni)
                acc[mi][ni] = wmma_f16(af[mi], bf[ni], acc[mi][ni]);
        __syncthreads();
    }
    const int half = lane >> 4, nl = lane & 15;
#pragma unroll
    for (int mi = 0; mi < 2; ++mi)
#pragma unroll
        for (int ni = 0; ni < 2; ++ni) {
            int ng = n0 + wn + ni * 16 + nl;
            float bb = bias[ng];
#pragma unroll
            for (int v = 0; v < 8; ++v) {
                int mg = m0 + wm + mi * 16 + v + 8 * half;
                out[(size_t)mg * DMODEL + ng] = acc[mi][ni][v] + bb;
            }
        }
}

// ---------------------------------------------------------------- launcher
extern "C" void kernel_launch(void* const* d_in, const int* in_sizes, int n_in,
                              void* d_out, int out_size, void* d_ws, size_t ws_size,
                              hipStream_t stream) {
    const float* q  = (const float*)d_in[0];
    const float* k  = (const float*)d_in[1];
    const float* v  = (const float*)d_in[2];
    // d_in[3] = mask : causality is applied analytically, mask not needed
    const float* wq = (const float*)d_in[4];
    const float* bq = (const float*)d_in[5];
    const float* wk = (const float*)d_in[6];
    const float* bk = (const float*)d_in[7];
    const float* wv = (const float*)d_in[8];
    const float* bv = (const float*)d_in[9];
    const float* wc = (const float*)d_in[10];
    const float* bc = (const float*)d_in[11];

    float* out  = (float*)d_out;                         // [4096][768]
    float* attn = out + (size_t)SEQ * DMODEL;            // [12][4096][4096]

    const size_t HEADS_ELEMS = (size_t)SEQ * DMODEL;     // == NH*SEQ*DH
    _Float16* qh  = (_Float16*)d_ws;
    _Float16* kh  = qh + HEADS_ELEMS;
    _Float16* vhp = kh + HEADS_ELEMS;
    _Float16* ctx = vhp + HEADS_ELEMS;

    dim3 gp(SEQ / 64, DMODEL / 64);
    qkv_proj_kernel<<<gp, 128, 0, stream>>>(q, wq, bq, qh);
    qkv_proj_kernel<<<gp, 128, 0, stream>>>(k, wk, bk, kh);
    qkv_proj_kernel<<<gp, 128, 0, stream>>>(v, wv, bv, vhp);

    dim3 gl(SEQ / 64, SEQ / 64, NH);
    logits_kernel<<<gl, 128, 0, stream>>>(qh, kh, attn);

    dim3 gs(SEQ, NH);
    softmax_kernel<<<gs, 256, 0, stream>>>(attn);

    dim3 gpv(SEQ / 64, 1, NH);
    pv_kernel<<<gpv, 128, 0, stream>>>(attn, vhp, ctx);

    out_proj_kernel<<<gp, 128, 0, stream>>>(ctx, wc, bc, out);
}
